// LSTM2Model_20048907338049
// MI455X (gfx1250) — matrix operand order repaired and checked by reading, not compiled, not run on hardware
//
#include <hip/hip_runtime.h>

typedef __bf16 bf16;
typedef __attribute__((ext_vector_type(16))) __bf16 v16bf;
typedef __attribute__((ext_vector_type(8)))  float  v8f;
typedef __attribute__((ext_vector_type(4)))  unsigned u32x4;
typedef __attribute__((ext_vector_type(8)))  int      i32x8;
typedef __attribute__((ext_vector_type(4)))  int      i32x4;

// ---------------------------------------------------------------------------
// WMMA helpers (bf16 A/B, f32 accumulate), CDNA5 16x16x32
// ---------------------------------------------------------------------------
__device__ inline v8f wmma_bf16(v16bf a, v16bf b, v8f c) {
  return __builtin_amdgcn_wmma_f32_16x16x32_bf16(false, a, false, b, (short)0, c,
                                                 false, false);
}

// A-matrix 16x32 bf16 fragment from row-major LDS [rows][stride]
__device__ inline v16bf load_frag_A(const bf16* base, int row0, int k0, int stride) {
  unsigned lane = threadIdx.x & 31u;
  int m = lane & 15, half = lane >> 4;
  const bf16* rp = base + (row0 + m) * stride;
  union { unsigned u[8]; v16bf v; } f;
#pragma unroll
  for (int vi = 0; vi < 8; ++vi) {
    int grp = vi >> 2, sub = vi & 3;
    int k = k0 + grp * 16 + half * 8 + sub * 2;
    f.u[vi] = *(const unsigned*)(rp + k);
  }
  return f.v;
}

// B-matrix 32x16 bf16 fragment; source row-major W[n][k] (B[k][n] = W[n][k])
__device__ inline v16bf load_frag_B(const bf16* base, int n0, int k0, int stride) {
  unsigned lane = threadIdx.x & 31u;
  int n = lane & 15, half = lane >> 4;
  const bf16* rp = base + (n0 + n) * stride;
  int kb = k0 + half * 8;
  union { unsigned u[8]; v16bf v; } f;
#pragma unroll
  for (int vi = 0; vi < 8; ++vi) f.u[vi] = *(const unsigned*)(rp + kb + (vi >> 2) * 16 + (vi & 3) * 2);
  return f.v;
}

// ---------------------------------------------------------------------------
// CDNA5 async/TDM data movement
// ---------------------------------------------------------------------------
// Async 16B global -> LDS copy (ASYNCcnt path). LDS offset = low 32 bits of the
// generic pointer (shared aperture base has zero low bits).
__device__ inline void async_b128(void* lds_ptr, const void* gptr) {
  unsigned off = (unsigned)(size_t)lds_ptr;
  asm volatile("global_load_async_to_lds_b128 %0, %1, off"
               :: "v"(off), "v"(gptr) : "memory");
}
__device__ inline void wait_async_le4() {
  asm volatile("s_wait_asynccnt 0x4" ::: "memory");
}
__device__ inline void wait_async_0() {
  asm volatile("s_wait_asynccnt 0x0" ::: "memory");
}

// TDM: one-instruction DMA of a 64x512 bf16 tile (row stride 512) into LDS with
// +16B pad after every 1024B row -> LDS row stride 520 bf16 (matches Hl layout).
__device__ inline void tdm_load_h64x512(unsigned lds_off, const void* gp) {
  unsigned long long ga = (unsigned long long)gp;
  u32x4 g0 = { 1u,                                    // count=1, user mode
               lds_off,                               // D#.lds_addr
               (unsigned)ga,                          // global_addr[31:0]
               ((unsigned)(ga >> 32) & 0x01FFFFFFu) | 0x80000000u }; // ga[56:32]|type=2
  i32x8 g1;
  g1[0] = (int)((1u << 16) |      // data_size = 2B
                (1u << 20) |      // pad_enable
                (7u << 22) |      // pad_interval = 256 dwords (one 1024B row)
                (3u << 25));      // pad_amount   = 4 dwords (16B)
  g1[1] = (int)(512u << 16);      // tensor_dim0 = 512  (bits 79:48)
  g1[2] = (int)(64u << 16);       // tensor_dim1 = 64   (bits 111:80)
  g1[3] = (int)(512u << 16);      // tile_dim0   = 512  (bits 127:112)
  g1[4] = 64;                     // tile_dim1 = 64, tile_dim2 = 0
  g1[5] = 512;                    // tensor_dim0_stride = 512 (elements)
  g1[6] = 0;
  g1[7] = 0;
  i32x4 gz = { 0, 0, 0, 0 };
#if defined(__clang_major__) && (__clang_major__ >= 23)
  i32x8 gz8 = { 0, 0, 0, 0, 0, 0, 0, 0 };
  __builtin_amdgcn_tensor_load_to_lds(g0, g1, gz, gz, gz8, 0);
#else
  __builtin_amdgcn_tensor_load_to_lds(g0, g1, gz, gz, 0);
#endif
}

// ---------------------------------------------------------------------------
// Prep kernels
// ---------------------------------------------------------------------------
__global__ void cvt_f32_to_bf16(const float* __restrict__ s, bf16* __restrict__ d) {
  int i = (blockIdx.x * 256 + threadIdx.x) * 4;
  d[i + 0] = (bf16)s[i + 0];
  d[i + 1] = (bf16)s[i + 1];
  d[i + 2] = (bf16)s[i + 2];
  d[i + 3] = (bf16)s[i + 3];
}

__global__ void lens_kernel(const int* __restrict__ x, int* __restrict__ lens) {
  int b = threadIdx.x;  // 64 threads
  int cnt = 0;
  for (int t = 0; t < 512; ++t) cnt += (x[b * 512 + t] == 0);
  lens[b] = 512 - cnt;
}

__global__ void embed_kernel(const int* __restrict__ x, const float* __restrict__ emb,
                             bf16* __restrict__ xe) {
  int r = blockIdx.x;  // r = t*64 + b
  int t = r >> 6, b = r & 63;
  int tok = x[b * 512 + t];
  const float* e = emb + (size_t)tok * 512;
  bf16* o = xe + (size_t)r * 512;
  int c = threadIdx.x * 4;  // 128 threads * 4
  float4 v = *(const float4*)(e + c);
  o[c + 0] = (bf16)v.x;
  o[c + 1] = (bf16)v.y;
  o[c + 2] = (bf16)v.z;
  o[c + 3] = (bf16)v.w;
}

// ---------------------------------------------------------------------------
// xproj GEMM: C[32768][2048] = A[M][512](bf16) @ W[2048][512]^T (bf16) + biases
// 128 threads (4 waves), 64x64 tile. K chunks (32) double-buffered in LDS via
// async global->LDS loads overlapped with WMMA on the other buffer.
// ---------------------------------------------------------------------------
__global__ __launch_bounds__(128) void gemm_xproj(
    const bf16* __restrict__ A, const bf16* __restrict__ W,
    const float* __restrict__ bih, const float* __restrict__ bhh,
    float* __restrict__ C) {
  __shared__ __align__(16) bf16 At[2][64 * 40];  // padded rows: 20 words, no conflicts
  __shared__ __align__(16) bf16 Bt[2][64 * 40];
  int n0 = blockIdx.x * 64;
  int m0 = blockIdx.y * 64;
  int tid = threadIdx.x;
  int wv = tid >> 5;
  unsigned lane = tid & 31u;
  int half = lane >> 4, nl = lane & 15;

  v8f acc[4];
#pragma unroll
  for (int j = 0; j < 4; ++j) acc[j] = (v8f){0, 0, 0, 0, 0, 0, 0, 0};

  // issue async stage for chunk kk into buffer s (4 x b128 per thread)
  auto stage = [&](int s, int k0) {
#pragma unroll
    for (int i = 0; i < 2; ++i) {
      int idx = tid * 2 + i;            // 0..255 : 64 rows x 4 x 16B
      int r = idx >> 2, c = (idx & 3) * 8;
      async_b128(&At[s][r * 40 + c], A + (size_t)(m0 + r) * 512 + k0 + c);
      async_b128(&Bt[s][r * 40 + c], W + (size_t)(n0 + r) * 512 + k0 + c);
    }
  };

  stage(0, 0);
  for (int kk = 0; kk < 16; ++kk) {
    int cur = kk & 1;
    if (kk < 15) {
      stage(cur ^ 1, (kk + 1) * 32);
      wait_async_le4();   // in-order: >=4 retired => chunk kk has landed
    } else {
      wait_async_0();
    }
    __syncthreads();
    v16bf af = load_frag_A(At[cur], wv * 16, 0, 40);
#pragma unroll
    for (int j = 0; j < 4; ++j) {
      v16bf bfr = load_frag_B(Bt[cur], j * 16, 0, 40);
      acc[j] = wmma_bf16(af, bfr, acc[j]);
    }
    __syncthreads();  // buffer cur is free to be overwritten next iteration
  }

#pragma unroll
  for (int j = 0; j < 4; ++j) {
    int col = n0 + j * 16 + nl;
    float bias = bih[col] + bhh[col];
#pragma unroll
    for (int r = 0; r < 8; ++r) {
      int row = m0 + wv * 16 + half * 8 + r;  // C/D layout: VGPR r -> M=r / M=r+8
      C[(size_t)row * 2048 + col] = acc[j][r] + bias;
    }
  }
}

// ---------------------------------------------------------------------------
// Persistent LSTM recurrence. Grid = 16 WGs x 256 threads (8 waves).
// WG w owns hidden units [w*32, w*32+32). W_hh slice (133KB bf16) resident in
// LDS all 512 steps. Per-step h state (64KB) pulled global->LDS by ONE TDM
// tensor_load_to_lds. Cross-WG sync: monotonic atomic counter + s_sleep spin.
// ---------------------------------------------------------------------------
__global__ __launch_bounds__(256) void lstm_scan(
    const float* __restrict__ xproj, const bf16* __restrict__ Whh,
    bf16* __restrict__ hA, bf16* __restrict__ hB, unsigned* __restrict__ cnt,
    bf16* __restrict__ hs_out, const int* __restrict__ lens,
    float* __restrict__ last) {
  extern __shared__ __align__(16) char smem[];
  bf16* Wl = (bf16*)smem;               // [128][520] bf16 = 133120 B
  bf16* Hl = Wl + 128 * 520;            // [64][520]  bf16 =  66560 B
  float* Gl = (float*)(Hl + 64 * 520);  // [64][132]  f32  =  33792 B
  float* Cl = Gl + 64 * 132;            // [64][32]   f32  =   8192 B

  const int tid = threadIdx.x;
  const int j0 = blockIdx.x * 32;
  const unsigned hl_off = (unsigned)(size_t)(void*)Hl;

  // Preload W_hh slice: local row l = gate*32 + jj <-> global row gate*512 + j0 + jj
  {
    unsigned* Wlu = (unsigned*)Wl;
    const unsigned* Wg = (const unsigned*)Whh;
    for (int i = 0; i < 128; ++i) {
      int idx = tid + 256 * i;
      int l = idx >> 8, cu = idx & 255;
      int gr = (l >> 5) * 512 + j0 + (l & 31);
      Wlu[l * 260 + cu] = Wg[(size_t)gr * 256 + cu];
    }
  }
  for (int i = 0; i < 8; ++i) Cl[tid + 256 * i] = 0.0f;

  const int ww = tid >> 5;
  const int mt = ww & 3;   // 4 m-tiles (64 batch rows)
  const int nh = ww >> 2;  // 2 halves of 8 n-tiles (128 gate cols)
  const unsigned lane = tid & 31u;
  const int half = lane >> 4, nl = lane & 15;
  __syncthreads();

  for (int t = 0; t < 512; ++t) {
    const bf16* hp = (t & 1) ? hB : hA;
    bf16* hn = (t & 1) ? hA : hB;

    // One TDM op stages h_prev (64x512 bf16, padded to 520/row) into LDS
    if (ww == 0) {
      tdm_load_h64x512(hl_off, hp);
      __builtin_amdgcn_s_wait_tensorcnt(0);
    }
    __syncthreads();

    // Gate GEMM: G[64][128] = h_prev[64][512] * Wl^T, all-WMMA from LDS
    v8f acc[4];
#pragma unroll
    for (int j = 0; j < 4; ++j) acc[j] = (v8f){0, 0, 0, 0, 0, 0, 0, 0};
    for (int kk = 0; kk < 16; ++kk) {
      v16bf af = load_frag_A(Hl, mt * 16, kk * 32, 520);
#pragma unroll
      for (int j = 0; j < 4; ++j) {
        v16bf bfr = load_frag_B(Wl, nh * 64 + j * 16, kk * 32, 520);
        acc[j] = wmma_bf16(af, bfr, acc[j]);
      }
    }
#pragma unroll
    for (int j = 0; j < 4; ++j) {
      int col = nh * 64 + j * 16 + nl;
#pragma unroll
      for (int r = 0; r < 8; ++r) {
        int row = mt * 16 + half * 8 + r;
        Gl[row * 132 + col] = acc[j][r];
      }
    }
    __syncthreads();

    // Elementwise gates: 64 b x 32 jj cells / 256 threads
#pragma unroll
    for (int i = 0; i < 8; ++i) {
      int idx = tid + 256 * i;
      int b = idx >> 5, jj = idx & 31;
      const float* xp = xproj + (size_t)(t * 64 + b) * 2048 + j0 + jj;
      float gi = Gl[b * 132 + jj]      + xp[0];
      float gf = Gl[b * 132 + 32 + jj] + xp[512];
      float gg = Gl[b * 132 + 64 + jj] + xp[1024];
      float go = Gl[b * 132 + 96 + jj] + xp[1536];
      gi = 1.0f / (1.0f + __expf(-gi));
      gf = 1.0f / (1.0f + __expf(-gf));
      gg = tanhf(gg);
      go = 1.0f / (1.0f + __expf(-go));
      float c = gf * Cl[idx] + gi * gg;
      Cl[idx] = c;
      float h = go * tanhf(c);
      hn[b * 512 + j0 + jj] = (bf16)h;
      if (hs_out) hs_out[(size_t)(t * 64 + b) * 512 + j0 + jj] = (bf16)h;
      if (last && t == lens[b] - 1) last[b * 512 + j0 + jj] = h;
    }
    __syncthreads();

    // Cross-workgroup step barrier (monotonic counter; 16 WGs co-resident)
    if (tid == 0) {
      __threadfence();
      atomicAdd(cnt, 1u);
      unsigned target = 16u * (unsigned)(t + 1);
      while (*(volatile unsigned*)cnt < target) { __builtin_amdgcn_s_sleep(1); }
      __threadfence();
    }
    __syncthreads();
  }
}

// ---------------------------------------------------------------------------
// Final projection: out[64][20] = last[64][512] @ W_out^T + b_out
// ---------------------------------------------------------------------------
__global__ void final_proj(const float* __restrict__ last, const float* __restrict__ Wout,
                           const float* __restrict__ bout, float* __restrict__ out) {
  int b = blockIdx.x;
  unsigned lane = threadIdx.x;  // 32
  float a[20];
#pragma unroll
  for (int tg = 0; tg < 20; ++tg) a[tg] = 0.0f;
  for (int k = lane; k < 512; k += 32) {
    float hv = last[b * 512 + k];
#pragma unroll
    for (int tg = 0; tg < 20; ++tg) a[tg] += hv * Wout[tg * 512 + k];
  }
#pragma unroll
  for (int tg = 0; tg < 20; ++tg) {
    float v = a[tg];
    for (int off = 16; off; off >>= 1) v += __shfl_down(v, off, 32);
    if (lane == 0) out[b * 20 + tg] = v + bout[tg];
  }
}

// ---------------------------------------------------------------------------
extern "C" void kernel_launch(void* const* d_in, const int* in_sizes, int n_in,
                              void* d_out, int out_size, void* d_ws, size_t ws_size,
                              hipStream_t stream) {
  const int*   x    = (const int*)d_in[0];
  const float* emb  = (const float*)d_in[1];
  const float* Wih0 = (const float*)d_in[2];
  const float* Whh0 = (const float*)d_in[3];
  const float* bih0 = (const float*)d_in[4];
  const float* bhh0 = (const float*)d_in[5];
  const float* Wih1 = (const float*)d_in[6];
  const float* Whh1 = (const float*)d_in[7];
  const float* bih1 = (const float*)d_in[8];
  const float* bhh1 = (const float*)d_in[9];
  const float* Wout = (const float*)d_in[10];
  const float* bout = (const float*)d_in[11];
  float* out = (float*)d_out;

  char* ws = (char*)d_ws;
  size_t off = 0;
  auto alloc = [&](size_t bytes) -> void* {
    void* p = ws + off;
    off = (off + bytes + 255) & ~(size_t)255;
    return p;
  };
  bf16*  xe    = (bf16*)alloc(512ull * 64 * 512 * 2);    // 32 MB
  bf16*  hs0   = (bf16*)alloc(512ull * 64 * 512 * 2);    // 32 MB
  float* xproj = (float*)alloc(512ull * 64 * 2048 * 4);  // 256 MB (reused per layer)
  bf16*  wih0b = (bf16*)alloc(2048ull * 512 * 2);
  bf16*  whh0b = (bf16*)alloc(2048ull * 512 * 2);
  bf16*  wih1b = (bf16*)alloc(2048ull * 512 * 2);
  bf16*  whh1b = (bf16*)alloc(2048ull * 512 * 2);
  bf16*  hA    = (bf16*)alloc(64 * 512 * 2);
  bf16*  hB    = (bf16*)alloc(64 * 512 * 2);
  float* lastb = (float*)alloc(64 * 512 * 4);
  int*   lens  = (int*)alloc(256);
  unsigned* cnt = (unsigned*)alloc(256);

  const size_t smem = (128 * 520 + 64 * 520) * 2 + (64 * 132 + 64 * 32) * 4;  // 241664 B
  hipFuncSetAttribute(reinterpret_cast<const void*>(lstm_scan),
                      hipFuncAttributeMaxDynamicSharedMemorySize, (int)smem);

  cvt_f32_to_bf16<<<1024, 256, 0, stream>>>(Wih0, wih0b);
  cvt_f32_to_bf16<<<1024, 256, 0, stream>>>(Whh0, whh0b);
  cvt_f32_to_bf16<<<1024, 256, 0, stream>>>(Wih1, wih1b);
  cvt_f32_to_bf16<<<1024, 256, 0, stream>>>(Whh1, whh1b);
  lens_kernel<<<1, 64, 0, stream>>>(x, lens);
  embed_kernel<<<32768, 128, 0, stream>>>(x, emb, xe);

  // Layer 0
  gemm_xproj<<<dim3(32, 512), 128, 0, stream>>>(xe, wih0b, bih0, bhh0, xproj);
  hipMemsetAsync(hA, 0, 64 * 512 * 2, stream);
  hipMemsetAsync(hB, 0, 64 * 512 * 2, stream);
  hipMemsetAsync(cnt, 0, 256, stream);
  lstm_scan<<<16, 256, smem, stream>>>(xproj, whh0b, hA, hB, cnt, hs0, lens, nullptr);

  // Layer 1
  gemm_xproj<<<dim3(32, 512), 128, 0, stream>>>(hs0, wih1b, bih1, bhh1, xproj);
  hipMemsetAsync(hA, 0, 64 * 512 * 2, stream);
  hipMemsetAsync(hB, 0, 64 * 512 * 2, stream);
  hipMemsetAsync(cnt, 0, 256, stream);
  lstm_scan<<<16, 256, smem, stream>>>(xproj, whh1b, hA, hB, cnt, nullptr, lens, lastb);

  final_proj<<<64, 32, 0, stream>>>(lastb, Wout, bout, out);
}
